// stack_witran_24034636988651
// MI455X (gfx1250) — matrix operand order repaired and checked by reading, not compiled, not run on hardware
//
#include <hip/hip_runtime.h>
#include <math.h>

typedef __attribute__((ext_vector_type(16))) _Float16 v16h;
typedef __attribute__((ext_vector_type(8)))  _Float16 v8h;
typedef __attribute__((ext_vector_type(8)))  float    v8f;
typedef __attribute__((ext_vector_type(4)))  int      v4i;

// pointer flavors for the gfx1250 async global->LDS builtin
typedef __attribute__((address_space(1))) v4i* gptr_v4i;
typedef __attribute__((address_space(3))) v4i* lptr_v4i;

#define HH     128   // hidden H
#define SIXH   768   // 6*H
#define KENC   384   // 3*H
#define RBNN   768   // COL * B * N
#define BNN    64    // B * N
#define ROWW   16
#define COLL   12
#define NSTEPS 27    // COL + ROW - 1
#define MR     12288 // B*N*L
#define CIN    64
#define NUMC   2

// gfx1250 async global->LDS path (guarded; falls back to plain copies)
#if defined(__has_builtin)
#if __has_builtin(__builtin_amdgcn_global_load_async_to_lds_b128) && \
    __has_builtin(__builtin_amdgcn_s_wait_asynccnt)
#define USE_ASYNC_LDS 1
#endif
#endif

__device__ __forceinline__ v8f wmma16(v16h a, v16h b, v8f c) {
  // v_wmma_f32_16x16x32_f16
  return __builtin_amdgcn_wmma_f32_16x16x32_f16(false, a, false, b, (short)0, c,
                                                false, false);
}

// A-tile (16x32 f16, MxK) per-lane load per ISA layout:
//   lanes 0-15 (hi=0): halves 0..7 -> K=kb+0..7,  halves 8..15 -> K=kb+16..23
//   lanes 16-31(hi=1): halves 0..7 -> K=kb+8..15, halves 8..15 -> K=kb+24..31
__device__ __forceinline__ v16h load_a16(const _Float16* rowp, int hi, int kb) {
  v8h lo = *(const v8h*)(rowp + kb + hi * 8);
  v8h hh = *(const v8h*)(rowp + kb + 16 + hi * 8);
  return __builtin_shufflevector(lo, hh, 0, 1, 2, 3, 4, 5, 6, 7,
                                 8, 9, 10, 11, 12, 13, 14, 15);
}

// B-tile: pre-packed contiguous 16 halves per lane.
__device__ __forceinline__ v16h load_b16(const _Float16* p) {
  v8h lo = *(const v8h*)(p);
  v8h hh = *(const v8h*)(p + 8);
  return __builtin_shufflevector(lo, hh, 0, 1, 2, 3, 4, 5, 6, 7,
                                 8, 9, 10, 11, 12, 13, 14, 15);
}

__device__ __forceinline__ float sigmoidf_(float x) {
  return 1.0f / (1.0f + __expf(-x));
}

// ---------------------------------------------------------------------------
// Pack W_enc (NUM,6H,3H) f32 -> f16 WMMA-B tiles.
// Layout: out[((c*48 + t)*12 + ks)*512 + lane*16 + i]
//   value = W[c][g=t*16+(lane&15)][k=ks*32 + (lane>>4)*8 + i + (i&8)]
// ---------------------------------------------------------------------------
__global__ void pack_wenc_kernel(const float* __restrict__ W,
                                 _Float16* __restrict__ out) {
  int idx = blockIdx.x * 256 + threadIdx.x;
  if (idx >= NUMC * 48 * 12 * 512) return;
  int i    = idx & 15;
  int lane = (idx >> 4) & 31;
  int rest = idx >> 9;
  int ks   = rest % 12;
  int tc   = rest / 12;
  int t    = tc % 48;
  int c    = tc / 48;
  int n = lane & 15, hi = lane >> 4;
  int g = t * 16 + n;
  int k = ks * 32 + hi * 8 + i + (i & 8);
  out[idx] = (_Float16)W[((size_t)c * SIXH + g) * KENC + k];
}

// fc3_w (128,64): B[k][h] = fc3_w[h][k]; 2 k-steps x 8 n-tiles
__global__ void pack_w3_kernel(const float* __restrict__ W,
                               _Float16* __restrict__ out) {
  int idx = blockIdx.x * 256 + threadIdx.x;
  if (idx >= 2 * 8 * 512) return;
  int i    = idx & 15;
  int lane = (idx >> 4) & 31;
  int rest = idx >> 9;           // ks*8 + t
  int ks = rest >> 3, t = rest & 7;
  int n = lane & 15, hi = lane >> 4;
  out[idx] = (_Float16)W[(size_t)(t * 16 + n) * CIN + ks * 32 + hi * 8 + i + (i & 8)];
}

// fc4_w (128,128): B[k][h] = fc4_w[h][k]; 4 k-steps x 8 n-tiles
__global__ void pack_w4_kernel(const float* __restrict__ W,
                               _Float16* __restrict__ out) {
  int idx = blockIdx.x * 256 + threadIdx.x;
  if (idx >= 4 * 8 * 512) return;
  int i    = idx & 15;
  int lane = (idx >> 4) & 31;
  int rest = idx >> 9;           // ks*8 + t
  int ks = rest >> 3, t = rest & 7;
  int n = lane & 15, hi = lane >> 4;
  out[idx] = (_Float16)W[(size_t)(t * 16 + n) * HH + ks * 32 + hi * 8 + i + (i & 8)];
}

// ---------------------------------------------------------------------------
// Fused fc3 (relu) + fc4: h16 = (relu(x@W3.T + b3))@W4.T + b4, stored f16.
// Block = 64 rows, 128 threads (4 waves); wave owns 16 rows x 128 cols.
// ---------------------------------------------------------------------------
__global__ void fc34_kernel(const float* __restrict__ x,
                            const float* __restrict__ b3,
                            const float* __restrict__ b4,
                            const _Float16* __restrict__ pw3,
                            const _Float16* __restrict__ pw4,
                            _Float16* __restrict__ h16) {
  __shared__ _Float16 tmp[64 * HH];   // 16 KB
  const int wave = threadIdx.x >> 5;
  const int lane = threadIdx.x & 31;
  const int n = lane & 15, hi = lane >> 4;
  const int m0 = blockIdx.x * 64 + wave * 16;

  // ---- GEMM1: 16x128, K=64, A = f32 x converted to f16 on the fly ----
  const float* xr = x + (size_t)(m0 + n) * CIN;
  v16h aA[2];
#pragma unroll
  for (int ks = 0; ks < 2; ++ks) {
    v8f f1 = *(const v8f*)(xr + ks * 32 + hi * 8);
    v8f f2 = *(const v8f*)(xr + ks * 32 + 16 + hi * 8);
    v16h a;
#pragma unroll
    for (int i = 0; i < 8; ++i) { a[i] = (_Float16)f1[i]; a[i + 8] = (_Float16)f2[i]; }
    aA[ks] = a;
  }
#pragma unroll
  for (int nt = 0; nt < 8; ++nt) {
    v8f acc;
    float bias = b3[nt * 16 + n];
#pragma unroll
    for (int e = 0; e < 8; ++e) acc[e] = bias;
#pragma unroll
    for (int ks = 0; ks < 2; ++ks)
      acc = wmma16(aA[ks], load_b16(pw3 + (size_t)(ks * 8 + nt) * 512 + lane * 16), acc);
#pragma unroll
    for (int e = 0; e < 8; ++e) {
      int m = e + 8 * hi;
      tmp[(wave * 16 + m) * HH + nt * 16 + n] = (_Float16)fmaxf(acc[e], 0.0f);
    }
  }
  __syncthreads();

  // ---- GEMM2: 16x128, K=128, A from LDS ----
  v8f acc2[8];
#pragma unroll
  for (int nt = 0; nt < 8; ++nt) {
    float bias = b4[nt * 16 + n];
#pragma unroll
    for (int e = 0; e < 8; ++e) acc2[nt][e] = bias;
  }
  const _Float16* trow = tmp + (size_t)(wave * 16 + n) * HH;
#pragma unroll
  for (int ks = 0; ks < 4; ++ks) {
    v16h a = load_a16(trow, hi, ks * 32);
#pragma unroll
    for (int nt = 0; nt < 8; ++nt)
      acc2[nt] = wmma16(a, load_b16(pw4 + (size_t)(ks * 8 + nt) * 512 + lane * 16), acc2[nt]);
  }
#pragma unroll
  for (int nt = 0; nt < 8; ++nt)
#pragma unroll
    for (int e = 0; e < 8; ++e) {
      int m = e + 8 * hi;
      h16[(size_t)(m0 + m) * HH + nt * 16 + n] = (_Float16)acc2[nt][e];
    }
}

// ---------------------------------------------------------------------------
// Hproj[c] = h16 @ Wx.T  (M=12288, N=768, K=128), f16 out.
// Grid (192, 12, 2), 128 threads; wave = 16 rows x 64 cols.
// ---------------------------------------------------------------------------
__global__ void xproj_kernel(const _Float16* __restrict__ h16,
                             const _Float16* __restrict__ pwenc,
                             _Float16* __restrict__ hproj) {
  const int wave = threadIdx.x >> 5;
  const int lane = threadIdx.x & 31;
  const int n = lane & 15, hi = lane >> 4;
  const int c = blockIdx.z;
  const int m0 = blockIdx.x * 64 + wave * 16;
  const int g0 = blockIdx.y * 64;

  v8f acc[4];
#pragma unroll
  for (int ti = 0; ti < 4; ++ti)
#pragma unroll
    for (int e = 0; e < 8; ++e) acc[ti][e] = 0.0f;

  const _Float16* arow = h16 + (size_t)(m0 + n) * HH;
#pragma unroll
  for (int ks = 0; ks < 4; ++ks) {
    v16h a = load_a16(arow, hi, ks * 32);
#pragma unroll
    for (int ti = 0; ti < 4; ++ti) {
      int t = blockIdx.y * 4 + ti;
      const _Float16* bp = pwenc + ((size_t)(c * 48 + t) * 12 + 8 + ks) * 512 + lane * 16;
      acc[ti] = wmma16(a, load_b16(bp), acc[ti]);
    }
  }
#pragma unroll
  for (int ti = 0; ti < 4; ++ti)
#pragma unroll
    for (int e = 0; e < 8; ++e) {
      int m = e + 8 * hi;
      hproj[((size_t)c * MR + (m0 + m)) * SIXH + g0 + ti * 16 + n] = (_Float16)acc[ti][e];
    }
}

// ---------------------------------------------------------------------------
// One recurrence step, fully fused: G = hrow@Wr.T + hcol@Wc.T + Gx(s) (+bias),
// gates, state update, h_col roll. Grid (12 slices, 8 H-tiles, 2 classes),
// 128 threads (4 waves); wave owns 16 rows x one 16-col tile of all 6 gates.
// All 4 waves consume the same 48 B-tiles -> stage them once into LDS via the
// CDNA5 async global->LDS path (ASYNCcnt), overlapping with C-init.
// ---------------------------------------------------------------------------
__global__ void step_kernel(int s,
                            const _Float16* __restrict__ hrow_in,
                            const _Float16* __restrict__ hcol_in,
                            _Float16* __restrict__ hrow_out,
                            _Float16* __restrict__ hcol_out,
                            const _Float16* __restrict__ hproj,
                            const _Float16* __restrict__ pwenc,
                            const float* __restrict__ Benc) {
  __shared__ _Float16 bsh[48 * 512];   // 48 KB: B-tiles [q*8+ks][512]
  const int tid  = threadIdx.x;
  const int wave = tid >> 5;
  const int lane = tid & 31;
  const int n = lane & 15, hi = lane >> 4;
  const int r  = blockIdx.x;           // slice id 0..11
  const int jt = blockIdx.y;           // H tile 0..7
  const int c  = blockIdx.z;
  const int j0 = jt * 16;
  const int n0 = r * 64 + wave * 16;   // state row base of this wave's tile

  // ---- stage the 48 shared B-tiles (16B chunks; 3072 chunks / 128 thr) ----
#pragma unroll
  for (int it = 0; it < 24; ++it) {
    int ch   = it * 128 + tid;         // chunk id
    int tile = ch >> 6;                // 64 chunks per 512-half tile
    int off  = (ch & 63) * 8;          // halves within tile
    int q    = tile >> 3;
    int ks   = tile & 7;
    const _Float16* src =
        pwenc + ((size_t)(c * 48 + q * 8 + jt) * 12 + ks) * 512 + off;
    _Float16* dst = bsh + tile * 512 + off;
#if defined(USE_ASYNC_LDS)
    __builtin_amdgcn_global_load_async_to_lds_b128((gptr_v4i)src, (lptr_v4i)dst,
                                                   0, 0);
#else
    *(v8h*)dst = *(const v8h*)src;
#endif
  }

  const _Float16* hri = hrow_in  + (size_t)c * RBNN * HH;
  const _Float16* hci = hcol_in  + (size_t)c * RBNN * HH;
  _Float16*       hro = hrow_out + (size_t)c * RBNN * HH;
  _Float16*       hco = hcol_out + (size_t)c * RBNN * HH;

  const bool haveBias = (r <= s) && (s < COLL);
  const bool haveX    = (s - r >= 0) && (s - r < ROWW);
  const int  lpos     = (s - r) * COLL + r;  // h row position within sample

  // C init: masked bias + gathered x-projection (overlaps async staging)
  v8f acc[6];
#pragma unroll
  for (int q = 0; q < 6; ++q) {
    int g = q * HH + j0 + n;
    float bias = haveBias ? Benc[(size_t)c * SIXH + g] : 0.0f;
#pragma unroll
    for (int e = 0; e < 8; ++e) {
      float v = bias;
      if (haveX) {
        int b = wave * 16 + e + 8 * hi;  // sample index 0..63
        v += (float)hproj[((size_t)c * MR + (size_t)b * 192 + lpos) * SIXH + g];
      }
      acc[q][e] = v;
    }
  }

#if defined(USE_ASYNC_LDS)
  __builtin_amdgcn_s_wait_asynccnt(0);
#endif
  __syncthreads();

  // K loop: ks 0..3 = h_row part, ks 4..7 = h_col part; B from LDS
#pragma unroll
  for (int ks = 0; ks < 8; ++ks) {
    const _Float16* st = (ks < 4) ? hri : hci;
    const _Float16* arow = st + (size_t)(n0 + n) * HH;
    v16h a = load_a16(arow, hi, (ks & 3) * 32);
#pragma unroll
    for (int q = 0; q < 6; ++q)
      acc[q] = wmma16(a, load_b16(bsh + ((q << 3) | ks) * 512 + lane * 16), acc[q]);
  }

  // Gates + state update + roll (all tiles share the C/D lane mapping)
#pragma unroll
  for (int e = 0; e < 8; ++e) {
    int m   = e + 8 * hi;
    int row = n0 + m;
    int col = j0 + n;
    float ur  = sigmoidf_(acc[0][e]);
    float orr = sigmoidf_(acc[1][e]);
    float uc  = sigmoidf_(acc[2][e]);
    float oc  = sigmoidf_(acc[3][e]);
    float ir  = tanhf(acc[4][e]);
    float ic  = tanhf(acc[5][e]);
    float hr_old = (float)hri[(size_t)row * HH + col];
    float hc_old = (float)hci[(size_t)row * HH + col];
    float hr_new = tanhf((1.0f - ur) * hr_old + ur * ir) * orr;
    float hc_new = tanhf((1.0f - uc) * hc_old + uc * ic) * oc;
    hro[(size_t)row * HH + col] = (_Float16)hr_new;
    int rrow = row + BNN; if (rrow >= RBNN) rrow -= RBNN;   // roll by BN
    hco[(size_t)rrow * HH + col] = (_Float16)hc_new;
  }
}

// ---------------------------------------------------------------------------
// Head: out[b*2+c] = 0.5*(col_last[b].f1w[c] + f1b[c]) + 0.5*(row_last[b].f2w[c] + f2b[c])
// row_last = final hrow rows 704..767; col_last = final (rolled) hcol rows 0..63.
// ---------------------------------------------------------------------------
__global__ void head_kernel(const _Float16* __restrict__ hrow_fin,
                            const _Float16* __restrict__ hcol_fin,
                            const float* __restrict__ f1w, const float* __restrict__ f1b,
                            const float* __restrict__ f2w, const float* __restrict__ f2b,
                            float* __restrict__ out) {
  int t = threadIdx.x;
  if (t >= 128) return;
  int b = t >> 1, c = t & 1;
  const _Float16* rl = hrow_fin + (size_t)c * RBNN * HH + (size_t)(11 * 64 + b) * HH;
  const _Float16* cl = hcol_fin + (size_t)c * RBNN * HH + (size_t)b * HH;
  float d1 = 0.0f, d2 = 0.0f;
#pragma unroll 4
  for (int k = 0; k < HH; ++k) {
    d1 += (float)cl[k] * f1w[c * HH + k];
    d2 += (float)rl[k] * f2w[c * HH + k];
  }
  out[b * 2 + c] = 0.5f * (d1 + f1b[c]) + 0.5f * (d2 + f2b[c]);
}

// ---------------------------------------------------------------------------
extern "C" void kernel_launch(void* const* d_in, const int* in_sizes, int n_in,
                              void* d_out, int out_size, void* d_ws, size_t ws_size,
                              hipStream_t stream) {
  const float* x    = (const float*)d_in[0];
  // d_in[1] = pad_mask (all true; unused by reference math)
  const float* fc3w = (const float*)d_in[2];
  const float* fc3b = (const float*)d_in[3];
  const float* fc4w = (const float*)d_in[4];
  const float* fc4b = (const float*)d_in[5];
  const float* Wenc = (const float*)d_in[6];
  const float* Benc = (const float*)d_in[7];
  const float* f1w  = (const float*)d_in[8];
  const float* f1b  = (const float*)d_in[9];
  const float* f2w  = (const float*)d_in[10];
  const float* f2b  = (const float*)d_in[11];
  float* out = (float*)d_out;

  char* base = (char*)d_ws;
  size_t o = 0;
  _Float16* h16   = (_Float16*)(base + o); o += (size_t)MR * HH * 2;            // 3.1 MB
  _Float16* pwenc = (_Float16*)(base + o); o += (size_t)NUMC * 48 * 12 * 1024;  // 1.2 MB
  _Float16* pw3   = (_Float16*)(base + o); o += (size_t)2 * 8 * 1024;
  _Float16* pw4   = (_Float16*)(base + o); o += (size_t)4 * 8 * 1024;
  _Float16* hproj = (_Float16*)(base + o); o += (size_t)NUMC * MR * SIXH * 2;   // 37.7 MB
  const size_t bufHalfs = (size_t)NUMC * RBNN * HH;      // one state buffer (both classes)
  _Float16* hrow  = (_Float16*)(base + o); o += bufHalfs * 2 * 2;               // dbl-buffered
  _Float16* hcol  = (_Float16*)(base + o); o += bufHalfs * 2 * 2;

  // zero initial state (both hrow/hcol double buffers; contiguous region)
  (void)hipMemsetAsync(hrow, 0, bufHalfs * 2 * 2 * 2, stream);

  pack_wenc_kernel<<<(NUMC * 48 * 12 * 512 + 255) / 256, 256, 0, stream>>>(Wenc, pwenc);
  pack_w3_kernel<<<(2 * 8 * 512 + 255) / 256, 256, 0, stream>>>(fc3w, pw3);
  pack_w4_kernel<<<(4 * 8 * 512 + 255) / 256, 256, 0, stream>>>(fc4w, pw4);

  fc34_kernel<<<MR / 64, 128, 0, stream>>>(x, fc3b, fc4b, pw3, pw4, h16);
  xproj_kernel<<<dim3(MR / 64, SIXH / 64, NUMC), 128, 0, stream>>>(h16, pwenc, hproj);

  for (int s = 0; s < NSTEPS; ++s) {
    _Float16* hri = hrow + (size_t)(s & 1) * bufHalfs;
    _Float16* hro = hrow + (size_t)((s & 1) ^ 1) * bufHalfs;
    _Float16* hci = hcol + (size_t)(s & 1) * bufHalfs;
    _Float16* hco = hcol + (size_t)((s & 1) ^ 1) * bufHalfs;
    step_kernel<<<dim3(COLL, 8, NUMC), 128, 0, stream>>>(s, hri, hci, hro, hco,
                                                         hproj, pwenc, Benc);
  }
  // NSTEPS = 27 (odd parity: last step wrote buffer 1)
  head_kernel<<<1, 128, 0, stream>>>(hrow + bufHalfs, hcol + bufHalfs,
                                     f1w, f1b, f2w, f2b, out);
}
